// EncoderImageAggr_9182640079365
// MI455X (gfx1250) — matrix operand order repaired and checked by reading, not compiled, not run on hardware
//
#include <hip/hip_runtime.h>
#include <math.h>

// ---------------------------------------------------------------------------
// Problem constants (from reference)
// ---------------------------------------------------------------------------
#define BB_   128
#define TT_   200
#define IMGD  2048
#define EMB   1024
#define HID   512            // EMBED/2
#define T1    100            // T/2
#define RSEL  10             // min(T//20, T//2)
#define LAGG  190            // 90 unm + 100 dst
#define DPE   32
#define DHID  32

// ---------------------------------------------------------------------------
// WMMA types (CDNA5 / gfx1250, wave32)
// ---------------------------------------------------------------------------
typedef __attribute__((ext_vector_type(16))) __bf16        v16bf;
typedef __attribute__((ext_vector_type(8)))  float         v8f;
typedef __attribute__((ext_vector_type(8)))  unsigned short v8us;

union ABu { v16bf bf; v8us us[2]; };

struct Frags { ABu a, b0, b1, b2, b3; };

__device__ __forceinline__ unsigned short f2bf(float f) {
    unsigned int u = __float_as_uint(f);
    u += 0x7FFFu + ((u >> 16) & 1u);        // round-to-nearest-even
    return (unsigned short)(u >> 16);
}

__device__ __forceinline__ void load_frags(Frags& f,
                                           const unsigned short* ap,
                                           const unsigned short* wp0, const unsigned short* wp1,
                                           const unsigned short* wp2, const unsigned short* wp3,
                                           int k0)
{
    f.a.us[0]  = *(const v8us*)(ap  + k0);
    f.a.us[1]  = *(const v8us*)(ap  + k0 + 8);
    f.b0.us[0] = *(const v8us*)(wp0 + k0);
    f.b0.us[1] = *(const v8us*)(wp0 + k0 + 8);
    f.b1.us[0] = *(const v8us*)(wp1 + k0);
    f.b1.us[1] = *(const v8us*)(wp1 + k0 + 8);
    f.b2.us[0] = *(const v8us*)(wp2 + k0);
    f.b2.us[1] = *(const v8us*)(wp2 + k0 + 8);
    f.b3.us[0] = *(const v8us*)(wp3 + k0);
    f.b3.us[1] = *(const v8us*)(wp3 + k0 + 8);
}

// ---------------------------------------------------------------------------
// fp32 -> bf16(bits) conversion
// ---------------------------------------------------------------------------
__global__ void cvt_bf16(const float* __restrict__ in, unsigned short* __restrict__ out, long long n) {
    long long i = (long long)blockIdx.x * blockDim.x + threadIdx.x;
    if (i < n) out[i] = f2bf(in[i]);
}

// ---------------------------------------------------------------------------
// Register-blocked bf16 WMMA GEMM: each wave computes a 16(M) x 64(N) C tile
// (4 accumulators, A fragment reused 4x), K-loop double-buffered so WMMA
// overlaps with the next fragment set's global loads.
// Requires M%16==0, N%64==0, K%64==0.
// mode 0: Out = acc + bias[col]                        (f32 store)
// mode 1: OutBf = bf16(relu(bn1(acc + bias)))          (bf16 store)
// mode 2: Out += bn2(acc + bias)                       (f32 RMW)
// ---------------------------------------------------------------------------
__global__ void __launch_bounds__(128, 1)
wmma_gemm_16x64(const unsigned short* __restrict__ A,
                const unsigned short* __restrict__ Wt,
                float* __restrict__ Out, unsigned short* __restrict__ OutBf,
                int M, int N, int K, int lda, int ldw, int ldc,
                const float* __restrict__ bias,
                const float* __restrict__ g,  const float* __restrict__ bb,
                const float* __restrict__ rm, const float* __restrict__ rv,
                int mode, int tilesN64, int numTiles)
{
    const int wave = threadIdx.x >> 5;
    const int lane = threadIdx.x & 31;
    const int tile = blockIdx.x * (blockDim.x >> 5) + wave;
    if (tile >= numTiles) return;                       // wave-uniform exit

    const int tm = tile / tilesN64, tn = tile % tilesN64;
    const int m0 = tm * 16, n0 = tn * 64;
    const int half = lane >> 4, l15 = lane & 15;

    // k order used here (contiguous, not the ISA table order; A and B match):
    //   A(16x32,bf16): lanes 0..15 -> K 0..15; lanes 16..31 -> K 16..31
    //   B(32x16,bf16): lanes 0..15 -> K 0..15; lanes 16..31 -> K 16..31
    const unsigned short* ap  = A  + (size_t)(m0 + l15)      * lda + half * 16;
    const unsigned short* wp0 = Wt + (size_t)(n0 + l15)      * ldw + half * 16;
    const unsigned short* wp1 = Wt + (size_t)(n0 + 16 + l15) * ldw + half * 16;
    const unsigned short* wp2 = Wt + (size_t)(n0 + 32 + l15) * ldw + half * 16;
    const unsigned short* wp3 = Wt + (size_t)(n0 + 48 + l15) * ldw + half * 16;

    v8f acc0 = {0.f,0.f,0.f,0.f,0.f,0.f,0.f,0.f};
    v8f acc1 = acc0, acc2 = acc0, acc3 = acc0;

    Frags f0, f1;
    load_frags(f0, ap, wp0, wp1, wp2, wp3, 0);          // preload set 0

    for (int k0 = 0; k0 < K; k0 += 64) {
        // stage set 1 (k0+32) while WMMA consumes set 0
        load_frags(f1, ap, wp0, wp1, wp2, wp3, k0 + 32);
        // speculative prefetch ~16 k-steps (1KB) ahead -> global_prefetch_b8
        __builtin_prefetch(ap  + k0 + 512, 0, 1);
        __builtin_prefetch(wp0 + k0 + 512, 0, 1);

        acc0 = __builtin_amdgcn_wmma_f32_16x16x32_bf16(false, f0.a.bf, false, f0.b0.bf, (short)0, acc0, false, false);
        acc1 = __builtin_amdgcn_wmma_f32_16x16x32_bf16(false, f0.a.bf, false, f0.b1.bf, (short)0, acc1, false, false);
        acc2 = __builtin_amdgcn_wmma_f32_16x16x32_bf16(false, f0.a.bf, false, f0.b2.bf, (short)0, acc2, false, false);
        acc3 = __builtin_amdgcn_wmma_f32_16x16x32_bf16(false, f0.a.bf, false, f0.b3.bf, (short)0, acc3, false, false);

        // stage set 0 for next iteration while WMMA consumes set 1
        if (k0 + 64 < K) load_frags(f0, ap, wp0, wp1, wp2, wp3, k0 + 64);

        acc0 = __builtin_amdgcn_wmma_f32_16x16x32_bf16(false, f1.a.bf, false, f1.b0.bf, (short)0, acc0, false, false);
        acc1 = __builtin_amdgcn_wmma_f32_16x16x32_bf16(false, f1.a.bf, false, f1.b1.bf, (short)0, acc1, false, false);
        acc2 = __builtin_amdgcn_wmma_f32_16x16x32_bf16(false, f1.a.bf, false, f1.b2.bf, (short)0, acc2, false, false);
        acc3 = __builtin_amdgcn_wmma_f32_16x16x32_bf16(false, f1.a.bf, false, f1.b3.bf, (short)0, acc3, false, false);
    }

    // Epilogue. C layout: lane<16 -> col=base+lane, rows m0+0..7 (vgpr j);
    //                     lane>=16 -> col=base+lane-16, rows m0+8..15.
    const int rbase = m0 + half * 8;
#pragma unroll
    for (int s = 0; s < 4; ++s) {
        const v8f acc = (s == 0) ? acc0 : (s == 1) ? acc1 : (s == 2) ? acc2 : acc3;
        const int col = n0 + s * 16 + l15;
        float bcol = bias ? bias[col] : 0.f;
        float gc = 0.f, bbc = 0.f, rmc = 0.f, inv = 0.f;
        if (mode == 1 || mode == 2) {
            gc = g[col]; bbc = bb[col]; rmc = rm[col];
            inv = rsqrtf(rv[col] + 1e-5f);
        }
#pragma unroll
        for (int j = 0; j < 8; ++j) {
            int row = rbase + j;
            float x = acc[j];
            if (mode == 0) {
                Out[(size_t)row * ldc + col] = x + bcol;
            } else if (mode == 1) {
                float y = gc * ((x + bcol) - rmc) * inv + bbc;
                y = fmaxf(y, 0.f);
                OutBf[(size_t)row * ldc + col] = f2bf(y);
            } else {
                float y = gc * ((x + bcol) - rmc) * inv + bbc;
                Out[(size_t)row * ldc + col] += y;
            }
        }
    }
}

// ---------------------------------------------------------------------------
// Generic (guarded, batched) bf16 WMMA GEMM, one wave per 16x16 C tile.
// Used for the padded 100x100x1024 batched scores GEMM.
// ---------------------------------------------------------------------------
__global__ void __launch_bounds__(128, 1)
wmma_gemm(const unsigned short* __restrict__ A,
          const unsigned short* __restrict__ Wt,
          float* __restrict__ Out,
          int M, int N, int K, int lda, int ldw, int ldc,
          long long strideA, long long strideW, long long strideC,
          int tilesN, int numTiles)
{
    const int wave = threadIdx.x >> 5;
    const int lane = threadIdx.x & 31;
    const int tile = blockIdx.x * (blockDim.x >> 5) + wave;
    if (tile >= numTiles) return;                       // wave-uniform exit

    const int bz = blockIdx.y;
    const unsigned short* Ab = A  + (size_t)bz * strideA;
    const unsigned short* Wb = Wt + (size_t)bz * strideW;

    const int tm = tile / tilesN, tn = tile % tilesN;
    const int m0 = tm * 16, n0 = tn * 16;
    const int half = lane >> 4, l15 = lane & 15;

    int arow = m0 + l15; if (arow >= M) arow = M - 1;   // clamp padded rows
    int wcol = n0 + l15; if (wcol >= N) wcol = N - 1;   // clamp padded cols

    const unsigned short* ap = Ab + (size_t)arow * lda + half * 16;
    const unsigned short* wp = Wb + (size_t)wcol * ldw + half * 16;

    v8f acc = {0.f,0.f,0.f,0.f,0.f,0.f,0.f,0.f};

    ABu a0, b0;
    a0.us[0] = *(const v8us*)(ap);      a0.us[1] = *(const v8us*)(ap + 8);
    b0.us[0] = *(const v8us*)(wp);      b0.us[1] = *(const v8us*)(wp + 8);

    for (int k0 = 0; k0 < K; k0 += 64) {
        ABu a1, b1;
        a1.us[0] = *(const v8us*)(ap + k0 + 32);
        a1.us[1] = *(const v8us*)(ap + k0 + 40);
        b1.us[0] = *(const v8us*)(wp + k0 + 32);
        b1.us[1] = *(const v8us*)(wp + k0 + 40);
        __builtin_prefetch(ap + k0 + 512, 0, 1);
        __builtin_prefetch(wp + k0 + 512, 0, 1);
        acc = __builtin_amdgcn_wmma_f32_16x16x32_bf16(false, a0.bf, false, b0.bf,
                                                      (short)0, acc, false, false);
        if (k0 + 64 < K) {
            a0.us[0] = *(const v8us*)(ap + k0 + 64);
            a0.us[1] = *(const v8us*)(ap + k0 + 72);
            b0.us[0] = *(const v8us*)(wp + k0 + 64);
            b0.us[1] = *(const v8us*)(wp + k0 + 72);
        }
        acc = __builtin_amdgcn_wmma_f32_16x16x32_bf16(false, a1.bf, false, b1.bf,
                                                      (short)0, acc, false, false);
    }

    const int col = n0 + l15;
    if (col >= N) return;                               // post-WMMA, per-lane OK
    const int rbase = m0 + half * 8;
    float* outp = Out + (size_t)bz * strideC;
#pragma unroll
    for (int j = 0; j < 8; ++j) {
        int row = rbase + j;
        if (row >= M) continue;
        outp[(size_t)row * ldc + col] = acc[j];
    }
}

// ---------------------------------------------------------------------------
// Row l2-norm + even/odd split into src/dst (f32 for gathers, bf16 for WMMA)
// ---------------------------------------------------------------------------
__global__ void l2norm_split(const float* __restrict__ feats,
                             float* __restrict__ srcf, float* __restrict__ dstf,
                             unsigned short* __restrict__ srcbf, unsigned short* __restrict__ dstbf)
{
    const int row = blockIdx.x;                 // b*T + t
    const int b = row / TT_, t = row % TT_;
    const float* fp = feats + (size_t)row * EMB;
    __shared__ float red[256];
    float ss = 0.f;
    for (int c = threadIdx.x; c < EMB; c += 256) { float v = fp[c]; ss += v * v; }
    red[threadIdx.x] = ss; __syncthreads();
    for (int o = 128; o > 0; o >>= 1) {
        if (threadIdx.x < o) red[threadIdx.x] += red[threadIdx.x + o];
        __syncthreads();
    }
    const float scale = 1.f / (sqrtf(red[0]) + 1e-8f);
    const size_t o = ((size_t)b * T1 + (t >> 1)) * EMB;
    float* fo = (t & 1) ? dstf : srcf;
    unsigned short* bo = (t & 1) ? dstbf : srcbf;
    for (int c = threadIdx.x; c < EMB; c += 256) {
        float v = fp[c] * scale;
        fo[o + c] = v; bo[o + c] = f2bf(v);
    }
}

// ---------------------------------------------------------------------------
// Per-batch: node_max/argmax over scores, stable descending argsort, counts
// ---------------------------------------------------------------------------
__global__ void aggr_meta(const float* __restrict__ scores,
                          int* __restrict__ orderG, int* __restrict__ nidxG,
                          float* __restrict__ cntG)
{
    const int b = blockIdx.x, tid = threadIdx.x;    // 128 threads
    __shared__ float nmax[T1]; __shared__ int nidx[T1];
    __shared__ int   ord[T1];  __shared__ float cnt[T1];
    const float* sb = scores + (size_t)b * T1 * T1;
    for (int s = tid; s < T1; s += blockDim.x) {
        float m = -3.0e38f; int mi = 0;
        for (int d = 0; d < T1; ++d) { float v = sb[s * T1 + d]; if (v > m) { m = v; mi = d; } }
        nmax[s] = m; nidx[s] = mi;
    }
    if (tid < T1) cnt[tid] = 0.f;
    __syncthreads();
    // stable argsort of -node_max (descending by value, ties by index asc)
    for (int s = tid; s < T1; s += blockDim.x) {
        float v = nmax[s]; int r = 0;
        for (int j = 0; j < T1; ++j) {
            float u = nmax[j];
            r += (u > v) || (u == v && j < s);
        }
        ord[r] = s;
    }
    __syncthreads();
    if (tid == 0) {
        for (int i = 0; i < RSEL; ++i) cnt[nidx[ord[i]]] += 1.f;
    }
    __syncthreads();
    for (int s = tid; s < T1; s += blockDim.x) {
        orderG[b * 128 + s] = ord[s];
        nidxG [b * 128 + s] = nidx[s];
        cntG  [b * 128 + s] = cnt[s];
    }
}

// ---------------------------------------------------------------------------
// Build aggregated features (B, 190, 1024): [unm(90) ; merged dst(100)]
// ---------------------------------------------------------------------------
__global__ void aggr_build(const float* __restrict__ srcf, const float* __restrict__ dstf,
                           const int* __restrict__ orderG, const int* __restrict__ nidxG,
                           const float* __restrict__ cntG, float* __restrict__ feats2)
{
    const int b = blockIdx.y, t = blockIdx.x;       // t in [0,190)
    const size_t ob = (size_t)b * T1 * EMB;
    float* outp = feats2 + ((size_t)b * LAGG + t) * EMB;
    if (t < LAGG - T1) {                            // 90 unmerged src rows
        int s = orderG[b * 128 + RSEL + t];
        const float* sp = srcf + ob + (size_t)s * EMB;
        for (int c = threadIdx.x; c < EMB; c += 256) outp[c] = sp[c];
    } else {
        int d = t - (LAGG - T1);
        float k = cntG[b * 128 + d];
        int msrc[RSEL]; int nm = 0;
        for (int i = 0; i < RSEL; ++i) {
            int si = orderG[b * 128 + i];
            if (nidxG[b * 128 + si] == d) msrc[nm++] = si;
        }
        float invc = 1.f / (1.f + k);
        for (int c = threadIdx.x; c < EMB; c += 256) {
            float v = dstf[ob + (size_t)d * EMB + c];
            for (int i = 0; i < nm; ++i) v += srcf[ob + (size_t)msrc[i] * EMB + c];
            outp[c] = v * invc;
        }
    }
}

// ---------------------------------------------------------------------------
// Bidirectional GRU over positional encoding + softmax weights. 1 wave.
// ---------------------------------------------------------------------------
__device__ void gru_scan_dir(const float* wih, const float* whh,
                             const float* bih, const float* bhh,
                             float* x, float* h, float* hsout, int j, int dir)
{
    h[j] = 0.f; __syncthreads();
    for (int s = 0; s < LAGG; ++s) {
        int t = dir ? (LAGG - 1 - s) : s;
        int i2 = (j >> 1) * 2;
        float div = expf(-(float)i2 * (logf(10000.f) / (float)DPE));
        float ang = (float)t * div;
        x[j] = (j & 1) ? cosf(ang) : sinf(ang);
        __syncthreads();
        float ir = bih[j], iz = bih[DHID + j], inn = bih[2 * DHID + j];
        float hr = bhh[j], hz = bhh[DHID + j], hn  = bhh[2 * DHID + j];
        for (int k = 0; k < DPE; ++k) {
            float xv = x[k], hv = h[k];
            ir  = fmaf(wih[j * DPE + k], xv, ir);
            iz  = fmaf(wih[(DHID + j) * DPE + k], xv, iz);
            inn = fmaf(wih[(2 * DHID + j) * DPE + k], xv, inn);
            hr  = fmaf(whh[j * DHID + k], hv, hr);
            hz  = fmaf(whh[(DHID + j) * DHID + k], hv, hz);
            hn  = fmaf(whh[(2 * DHID + j) * DHID + k], hv, hn);
        }
        float r = 1.f / (1.f + expf(-(ir + hr)));
        float z = 1.f / (1.f + expf(-(iz + hz)));
        float n = tanhf(inn + r * hn);
        float hnew = (1.f - z) * n + z * h[j];
        __syncthreads();
        h[j] = hnew; __syncthreads();
        hsout[t * DHID + j] = hnew;
    }
}

__global__ void gru_select(const float* __restrict__ wihf, const float* __restrict__ whhf,
                           const float* __restrict__ bihf, const float* __restrict__ bhhf,
                           const float* __restrict__ wihb, const float* __restrict__ whhb,
                           const float* __restrict__ bihb, const float* __restrict__ bhhb,
                           const float* __restrict__ linw,
                           float* __restrict__ hsbuf, float* __restrict__ wout)
{
    const int j = threadIdx.x;                      // 32 threads
    __shared__ float x[DPE], h[DHID], sc[LAGG];
    gru_scan_dir(wihf, whhf, bihf, bhhf, x, h, hsbuf, j, 0);
    gru_scan_dir(wihb, whhb, bihb, bhhb, x, h, hsbuf + LAGG * DHID, j, 1);
    __syncthreads();
    for (int t = j; t < LAGG; t += 32) {
        float d = 0.f;
        for (int k = 0; k < DHID; ++k)
            d += 0.5f * (hsbuf[t * DHID + k] + hsbuf[(LAGG + t) * DHID + k]) * linw[k];
        sc[t] = d * 10.f;                           // /0.1
    }
    __syncthreads();
    if (j == 0) {
        float mx = -3.0e38f;
        for (int t = 0; t < LAGG; ++t) mx = fmaxf(mx, sc[t]);
        float s = 0.f;
        for (int t = 0; t < LAGG; ++t) { float e = expf(sc[t] - mx); sc[t] = e; s += e; }
        for (int t = 0; t < LAGG; ++t) wout[t] = sc[t] / s;
    }
}

// ---------------------------------------------------------------------------
// srt = sort desc along T then pooled = sum(srt * w): computed via ranks.
// ---------------------------------------------------------------------------
__global__ void rankpool(const float* __restrict__ feats2, const float* __restrict__ wvec,
                         float* __restrict__ pooled)
{
    const int b = blockIdx.y;
    const int c = blockIdx.x * 64 + threadIdx.x;
    __shared__ float tile[LAGG * 64];
    __shared__ float wsh[LAGG];
    for (int t = threadIdx.x; t < LAGG; t += 64) wsh[t] = wvec[t];
    for (int t = 0; t < LAGG; ++t)
        tile[t * 64 + threadIdx.x] = feats2[((size_t)b * LAGG + t) * EMB + c];
    __syncthreads();
    float acc = 0.f;
    for (int t = 0; t < LAGG; ++t) {
        float xv = tile[t * 64 + threadIdx.x];
        int rnk = 0;
        for (int u = 0; u < LAGG; ++u) {
            float xu = tile[u * 64 + threadIdx.x];
            rnk += (xu > xv) || (xu == xv && u > t);    // reversed-stable desc
        }
        acc += xv * wsh[rnk];
    }
    pooled[(size_t)b * EMB + c] = acc;
}

__global__ void final_norm(const float* __restrict__ pooled, float* __restrict__ out)
{
    const int b = blockIdx.x;
    __shared__ float red[256];
    float ss = 0.f;
    for (int c = threadIdx.x; c < EMB; c += 256) { float v = pooled[(size_t)b * EMB + c]; ss += v * v; }
    red[threadIdx.x] = ss; __syncthreads();
    for (int o = 128; o > 0; o >>= 1) {
        if (threadIdx.x < o) red[threadIdx.x] += red[threadIdx.x + o];
        __syncthreads();
    }
    float scale = 1.f / (sqrtf(red[0]) + 1e-8f);
    for (int c = threadIdx.x; c < EMB; c += 256)
        out[(size_t)b * EMB + c] = pooled[(size_t)b * EMB + c] * scale;
}

// ---------------------------------------------------------------------------
// Host launcher
// ---------------------------------------------------------------------------
extern "C" void kernel_launch(void* const* d_in, const int* in_sizes, int n_in,
                              void* d_out, int out_size, void* d_ws, size_t ws_size,
                              hipStream_t stream)
{
    (void)in_sizes; (void)n_in; (void)out_size; (void)ws_size;
    const float* images = (const float*)d_in[0];
    const float* fc_w   = (const float*)d_in[2];
    const float* fc_b   = (const float*)d_in[3];
    const float* w1     = (const float*)d_in[4];
    const float* b1     = (const float*)d_in[5];
    const float* bn1g   = (const float*)d_in[6];
    const float* bn1b   = (const float*)d_in[7];
    const float* bn1rm  = (const float*)d_in[8];
    const float* bn1rv  = (const float*)d_in[9];
    const float* w2     = (const float*)d_in[10];
    const float* b2     = (const float*)d_in[11];
    const float* bn2g   = (const float*)d_in[12];
    const float* bn2b   = (const float*)d_in[13];
    const float* bn2rm  = (const float*)d_in[14];
    const float* bn2rv  = (const float*)d_in[15];
    const float* gwihf  = (const float*)d_in[16];
    const float* gwhhf  = (const float*)d_in[17];
    const float* gbihf  = (const float*)d_in[18];
    const float* gbhhf  = (const float*)d_in[19];
    const float* gwihb  = (const float*)d_in[20];
    const float* gwhhb  = (const float*)d_in[21];
    const float* gbihb  = (const float*)d_in[22];
    const float* gbhhb  = (const float*)d_in[23];
    const float* linw   = (const float*)d_in[24];
    float* out = (float*)d_out;

    // workspace layout (bytes); IMG_BF region is later reused for srcf/dstf,
    // FEATS region for feats2, HBF region for srcbf.
    char* ws = (char*)d_ws;
    const long long MROWS = (long long)BB_ * TT_;              // 25600
    unsigned short* img_bf = (unsigned short*)(ws + 0);        // 104,857,600 B
    unsigned short* wfc_bf = (unsigned short*)(ws + 104857600LL);
    unsigned short* w1_bf  = (unsigned short*)(ws + 109051904LL);
    unsigned short* w2_bf  = (unsigned short*)(ws + 111149056LL);
    float*          feats  = (float*)        (ws + 112197632LL);
    unsigned short* h_bf   = (unsigned short*)(ws + 217055232LL);
    float*          srcf   = (float*)        (ws + 0);            // alias img_bf
    float*          dstf   = (float*)        (ws + 52428800LL);
    unsigned short* srcbf  = (unsigned short*)(ws + 217055232LL); // alias h_bf
    unsigned short* dstbf  = (unsigned short*)(ws + 243269632LL);
    float*          scores = (float*)        (ws + 269484032LL);
    int*            orderG = (int*)          (ws + 274604032LL);
    int*            nidxG  = (int*)          (ws + 274669568LL);
    float*          cntG   = (float*)        (ws + 274735104LL);
    float*          hsbuf  = (float*)        (ws + 274800640LL);
    float*          wvec   = (float*)        (ws + 274849280LL);
    float*          pooled = (float*)        (ws + 274850048LL);
    float*          feats2 = (float*)        (ws + 112197632LL); // alias feats

    // 1) bf16 conversions
    {
        long long n = MROWS * IMGD;
        cvt_bf16<<<dim3((unsigned)((n + 255) / 256)), 256, 0, stream>>>(images, img_bf, n);
        n = (long long)EMB * IMGD;
        cvt_bf16<<<dim3((unsigned)((n + 255) / 256)), 256, 0, stream>>>(fc_w, wfc_bf, n);
        n = (long long)HID * IMGD;
        cvt_bf16<<<dim3((unsigned)((n + 255) / 256)), 256, 0, stream>>>(w1, w1_bf, n);
        n = (long long)EMB * HID;
        cvt_bf16<<<dim3((unsigned)((n + 255) / 256)), 256, 0, stream>>>(w2, w2_bf, n);
    }

    // 2) feats = images @ fc_w^T + fc_b            (mode 0)
    {
        int tilesN64 = EMB / 64, numTiles = (int)(MROWS / 16) * tilesN64; // 25600
        wmma_gemm_16x64<<<dim3(numTiles / 4, 1), 128, 0, stream>>>(
            img_bf, wfc_bf, feats, nullptr,
            (int)MROWS, EMB, IMGD, IMGD, IMGD, EMB,
            fc_b, nullptr, nullptr, nullptr, nullptr, 0, tilesN64, numTiles);
    }
    // 3) h_bf = bf16(relu(bn1(images @ w1^T + b1)))  (mode 1)
    {
        int tilesN64 = HID / 64, numTiles = (int)(MROWS / 16) * tilesN64; // 12800
        wmma_gemm_16x64<<<dim3(numTiles / 4, 1), 128, 0, stream>>>(
            img_bf, w1_bf, nullptr, h_bf,
            (int)MROWS, HID, IMGD, IMGD, IMGD, HID,
            b1, bn1g, bn1b, bn1rm, bn1rv, 1, tilesN64, numTiles);
    }
    // 4) feats += bn2(h_bf @ w2^T + b2)            (mode 2)
    {
        int tilesN64 = EMB / 64, numTiles = (int)(MROWS / 16) * tilesN64; // 25600
        wmma_gemm_16x64<<<dim3(numTiles / 4, 1), 128, 0, stream>>>(
            h_bf, w2_bf, feats, nullptr,
            (int)MROWS, EMB, HID, HID, HID, EMB,
            b2, bn2g, bn2b, bn2rm, bn2rv, 2, tilesN64, numTiles);
    }

    // 5) l2norm + even/odd split
    l2norm_split<<<dim3((unsigned)MROWS), 256, 0, stream>>>(feats, srcf, dstf, srcbf, dstbf);

    // 6) scores[b] = src_bf[b] @ dst_bf[b]^T  (batched, M=N=100 padded to 112)
    {
        int tilesN = 7, numTiles = 7 * 7;
        wmma_gemm<<<dim3((numTiles + 3) / 4, BB_), 128, 0, stream>>>(
            srcbf, dstbf, scores,
            T1, T1, EMB, EMB, EMB, T1,
            (long long)T1 * EMB, (long long)T1 * EMB, (long long)T1 * T1,
            tilesN, numTiles);
    }

    // 7) per-batch merge metadata + aggregated features (B,190,1024)
    aggr_meta<<<dim3(BB_), 128, 0, stream>>>(scores, orderG, nidxG, cntG);
    aggr_build<<<dim3(LAGG, BB_), 256, 0, stream>>>(srcf, dstf, orderG, nidxG, cntG, feats2);

    // 8) bidirectional GRU over PE + softmax -> weights w[190]
    gru_select<<<dim3(1), 32, 0, stream>>>(gwihf, gwhhf, gbihf, gbhhf,
                                           gwihb, gwhhb, gbihb, gbhhb,
                                           linw, hsbuf, wvec);

    // 9) descending-sort weighted pooling + final l2norm
    rankpool<<<dim3(EMB / 64, BB_), 64, 0, stream>>>(feats2, wvec, pooled);
    final_norm<<<dim3(BB_), 256, 0, stream>>>(pooled, out);
}
